// Net_40080634806568
// MI455X (gfx1250) — matrix operand order repaired and checked by reading, not compiled, not run on hardware
//
#include <hip/hip_runtime.h>
#include <hip/hip_bf16.h>

typedef __attribute__((ext_vector_type(16))) _Float16 v16h;
typedef __attribute__((ext_vector_type(8)))  _Float16 h8;
typedef __attribute__((ext_vector_type(8)))  float    v8f;

#define D_IN   784
#define KPAD   800        // 784 padded to multiple of 32
#define H      2048
#define BATCH  512
#define NLAB   10
#define ROWS   (NLAB * BATCH)   // 5120
#define EPS_N  1e-4f

// ---------------- weight convert f32 -> f16, with optional column pad ----------------
__global__ void wconv_kernel(const float* __restrict__ src, _Float16* __restrict__ dst,
                             int rows, int cols, int ldd) {
    int idx = blockIdx.x * blockDim.x + threadIdx.x;
    int total = rows * ldd;
    if (idx >= total) return;
    int r = idx / ldd, c = idx - r * ldd;
    dst[idx] = (c < cols) ? (_Float16)src[r * cols + c] : (_Float16)0.0f;
}

// ---------------- global max(x), two-pass ----------------
__global__ void max1_kernel(const float* __restrict__ x, float* __restrict__ part, int n) {
    __shared__ float sm[256];
    float m = -3.4e38f;
    for (int i = blockIdx.x * blockDim.x + threadIdx.x; i < n; i += gridDim.x * blockDim.x)
        m = fmaxf(m, x[i]);
    sm[threadIdx.x] = m; __syncthreads();
    for (int s = 128; s > 0; s >>= 1) {
        if (threadIdx.x < s) sm[threadIdx.x] = fmaxf(sm[threadIdx.x], sm[threadIdx.x + s]);
        __syncthreads();
    }
    if (threadIdx.x == 0) part[blockIdx.x] = sm[0];
}
__global__ void max2_kernel(const float* __restrict__ part, float* __restrict__ outv, int n) {
    __shared__ float sm[256];
    float m = -3.4e38f;
    for (int i = threadIdx.x; i < n; i += 256) m = fmaxf(m, part[i]);
    sm[threadIdx.x] = m; __syncthreads();
    for (int s = 128; s > 0; s >>= 1) {
        if (threadIdx.x < s) sm[threadIdx.x] = fmaxf(sm[threadIdx.x], sm[threadIdx.x + s]);
        __syncthreads();
    }
    if (threadIdx.x == 0) outv[0] = sm[0];
}

// ---------------- overlay label + L2-normalize row + pad to KPAD, f16 out ----------------
__global__ void hx_kernel(const float* __restrict__ x, const float* __restrict__ maxv_p,
                          _Float16* __restrict__ dst) {
    int row = blockIdx.x;            // l*512 + b
    int l = row >> 9, b = row & 511;
    float maxv = maxv_p[0];
    __shared__ float sm[256];
    float ss = 0.f;
    for (int c = threadIdx.x; c < D_IN; c += 256) {
        float v = (c < NLAB) ? ((c == l) ? maxv : 0.f) : x[b * D_IN + c];
        ss += v * v;
    }
    sm[threadIdx.x] = ss; __syncthreads();
    for (int s = 128; s > 0; s >>= 1) {
        if (threadIdx.x < s) sm[threadIdx.x] += sm[threadIdx.x + s];
        __syncthreads();
    }
    float scale = 1.0f / (sqrtf(sm[0]) + EPS_N);
    for (int c = threadIdx.x; c < KPAD; c += 256) {
        float v = 0.f;
        if (c < D_IN) v = (c < NLAB) ? ((c == l) ? maxv : 0.f) : x[b * D_IN + c];
        dst[(size_t)row * KPAD + c] = (_Float16)(v * scale);
    }
}

// ---------------- L2-normalize f32 state row -> f16 ----------------
__global__ void norm_kernel(const float* __restrict__ s, _Float16* __restrict__ dst) {
    int row = blockIdx.x;
    __shared__ float sm[256];
    const float* p = s + (size_t)row * H;
    float ss = 0.f;
    for (int c = threadIdx.x; c < H; c += 256) { float v = p[c]; ss += v * v; }
    sm[threadIdx.x] = ss; __syncthreads();
    for (int st = 128; st > 0; st >>= 1) {
        if (threadIdx.x < st) sm[threadIdx.x] += sm[threadIdx.x + st];
        __syncthreads();
    }
    float scale = 1.0f / (sqrtf(sm[0]) + EPS_N);
    _Float16* q = dst + (size_t)row * H;
    for (int c = threadIdx.x; c < H; c += 256) q[c] = (_Float16)(p[c] * scale);
}

// ---------------- WMMA fragment loads (wave32 layouts per CDNA5 ISA 7.12.2) ----------------
// A (16x32 f16): lane m=lane&15, half=lane>>4 -> elems 0..7 = K(half*8+0..7), 8..15 = K(16+half*8+0..7)
static __device__ __forceinline__ v16h load_a_frag(const _Float16* p, int half) {
    h8 lo = *(const h8*)(p + half * 8);
    h8 hi = *(const h8*)(p + 16 + half * 8);
    v16h r;
#pragma unroll
    for (int i = 0; i < 8; ++i) { r[i] = lo[i]; r[i + 8] = hi[i]; }
    return r;
}
// B (32x16 f16): lane n=lane&15, half=lane>>4 -> elems 0..7 = K(half*8+0..7), 8..15 = K(16+half*8+0..7)
// B[k][n] == W[n][k] so column n of B is a contiguous row of W.
static __device__ __forceinline__ v16h load_b_frag(const _Float16* p, int half) {
    h8 lo = *(const h8*)(p + half * 8);
    h8 hi = *(const h8*)(p + 16 + half * 8);
    v16h r;
#pragma unroll
    for (int i = 0; i < 8; ++i) { r[i] = lo[i]; r[i + 8] = hi[i]; }
    return r;
}

// one GEMM branch: acc[2][4] += A[m0:m0+32, :K] x W[n0:n0+64, :K]^T
// 32x64 per-wave tile: 2 A frags + 4 B frags per K-step -> 8 WMMAs
// (B reused across both M-halves, A reused across 4 N-tiles: 0.75KB L2 traffic/WMMA)
static __device__ __forceinline__ void mm_strip(const _Float16* __restrict__ A,
                                                const _Float16* __restrict__ W,
                                                int K, int m0, int n0,
                                                int lane16, int half, v8f acc[2][4]) {
    const _Float16* arow0 = A + (size_t)(m0 + lane16) * K;
    const _Float16* arow1 = A + (size_t)(m0 + 16 + lane16) * K;
    const _Float16* w0    = W + (size_t)(n0 + lane16) * K;
    for (int k = 0; k < K; k += 32) {
        v16h a0 = load_a_frag(arow0 + k, half);
        v16h a1 = load_a_frag(arow1 + k, half);
#pragma unroll
        for (int t = 0; t < 4; ++t) {
            v16h bf = load_b_frag(w0 + (size_t)t * 16 * K + k, half);
            acc[0][t] = __builtin_amdgcn_wmma_f32_16x16x32_f16(
                false, a0, false, bf, (short)0, acc[0][t], false, false);
            acc[1][t] = __builtin_amdgcn_wmma_f32_16x16x32_f16(
                false, a1, false, bf, (short)0, acc[1][t], false, false);
        }
    }
}

// Out = 0.7*(relu(Ap Wp^T + bp) + relu(Aq Wq^T + bq)) + 0.3*relu(As Ws^T + bs)
// K==0 disables a branch (acts as relu(bias), matching zero/width-0 inputs).
__global__ __launch_bounds__(256) void meta_gemm_kernel(
    const _Float16* __restrict__ Ap, const _Float16* __restrict__ Wp, const float* __restrict__ bp, int Kp,
    const _Float16* __restrict__ Aq, const _Float16* __restrict__ Wq, const float* __restrict__ bq, int Kq,
    const _Float16* __restrict__ As, const _Float16* __restrict__ Ws, const float* __restrict__ bs, int Ks,
    float* __restrict__ Out)
{
    const int lane   = threadIdx.x & 31;
    const int wave   = threadIdx.x >> 5;
    const int lane16 = lane & 15;
    const int half   = lane >> 4;
    const int m0 = (blockIdx.y * 8 + wave) * 32;
    const int n0 = blockIdx.x * 64;

    float res[2][4][8];
    v8f acc[2][4];

    // ---- pre branch (x0.7) ----
#pragma unroll
    for (int u = 0; u < 2; ++u)
#pragma unroll
        for (int t = 0; t < 4; ++t)
#pragma unroll
            for (int i = 0; i < 8; ++i) acc[u][t][i] = 0.f;
    if (Kp > 0) mm_strip(Ap, Wp, Kp, m0, n0, lane16, half, acc);
#pragma unroll
    for (int t = 0; t < 4; ++t) {
        float bv = bp[n0 + t * 16 + lane16];
#pragma unroll
        for (int u = 0; u < 2; ++u)
#pragma unroll
            for (int r = 0; r < 8; ++r) res[u][t][r] = 0.7f * fmaxf(acc[u][t][r] + bv, 0.f);
    }

    // ---- post branch (x0.7) ----
#pragma unroll
    for (int u = 0; u < 2; ++u)
#pragma unroll
        for (int t = 0; t < 4; ++t)
#pragma unroll
            for (int i = 0; i < 8; ++i) acc[u][t][i] = 0.f;
    if (Kq > 0) mm_strip(Aq, Wq, Kq, m0, n0, lane16, half, acc);
#pragma unroll
    for (int t = 0; t < 4; ++t) {
        float bv = bq[n0 + t * 16 + lane16];
#pragma unroll
        for (int u = 0; u < 2; ++u)
#pragma unroll
            for (int r = 0; r < 8; ++r) res[u][t][r] += 0.7f * fmaxf(acc[u][t][r] + bv, 0.f);
    }

    // ---- self branch (x0.3) ----
#pragma unroll
    for (int u = 0; u < 2; ++u)
#pragma unroll
        for (int t = 0; t < 4; ++t)
#pragma unroll
            for (int i = 0; i < 8; ++i) acc[u][t][i] = 0.f;
    if (Ks > 0) mm_strip(As, Ws, Ks, m0, n0, lane16, half, acc);
#pragma unroll
    for (int t = 0; t < 4; ++t) {
        float bv = bs[n0 + t * 16 + lane16];
#pragma unroll
        for (int u = 0; u < 2; ++u)
#pragma unroll
            for (int r = 0; r < 8; ++r) res[u][t][r] += 0.3f * fmaxf(acc[u][t][r] + bv, 0.f);
    }

    // ---- store (C/D layout: VGPR r -> M = base + r + half*8, N = n0 + t*16 + lane16) ----
#pragma unroll
    for (int u = 0; u < 2; ++u) {
#pragma unroll
        for (int t = 0; t < 4; ++t) {
            int col = n0 + t * 16 + lane16;
#pragma unroll
            for (int r = 0; r < 8; ++r) {
                int row = m0 + u * 16 + r + half * 8;
                Out[(size_t)row * H + col] = res[u][t][r];
            }
        }
    }
}

// ---------------- goodness: out[b,l] = (sum s1^2 + sum s2^2 + sum s3^2)/H ----------------
__global__ void goodness_kernel(const float* __restrict__ s1, const float* __restrict__ s2,
                                const float* __restrict__ s3, float* __restrict__ out) {
    int row = blockIdx.x;      // l*512 + b
    int l = row >> 9, b = row & 511;
    __shared__ float sm[256];
    size_t base = (size_t)row * H;
    float ss = 0.f;
    for (int c = threadIdx.x; c < H; c += 256) {
        float a = s1[base + c], d = s2[base + c], e = s3[base + c];
        ss += a * a + d * d + e * e;
    }
    sm[threadIdx.x] = ss; __syncthreads();
    for (int s = 128; s > 0; s >>= 1) {
        if (threadIdx.x < s) sm[threadIdx.x] += sm[threadIdx.x + s];
        __syncthreads();
    }
    if (threadIdx.x == 0) out[b * NLAB + l] = sm[0] * (1.0f / (float)H);
}

extern "C" void kernel_launch(void* const* d_in, const int* in_sizes, int n_in,
                              void* d_out, int out_size, void* d_ws, size_t ws_size,
                              hipStream_t stream) {
    (void)in_sizes; (void)n_in; (void)out_size; (void)ws_size;
    const float* x       = (const float*)d_in[0];
    const float* w1_pre  = (const float*)d_in[1];
    const float* b1_pre  = (const float*)d_in[2];
    const float* w1_post = (const float*)d_in[3];
    const float* b1_post = (const float*)d_in[4];
    const float* w1_self = (const float*)d_in[5];
    const float* b1_self = (const float*)d_in[6];
    const float* w2_pre  = (const float*)d_in[7];
    const float* b2_pre  = (const float*)d_in[8];
    const float* w2_post = (const float*)d_in[9];
    const float* b2_post = (const float*)d_in[10];
    const float* w2_self = (const float*)d_in[11];
    const float* b2_self = (const float*)d_in[12];
    const float* w3_pre  = (const float*)d_in[13];
    const float* b3_pre  = (const float*)d_in[14];
    /* d_in[15] = w3_post, width 0, unused */
    const float* b3_post = (const float*)d_in[16];
    const float* w3_self = (const float*)d_in[17];
    const float* b3_self = (const float*)d_in[18];
    float* out = (float*)d_out;

    char* ws = (char*)d_ws;
    size_t off = 0;
    auto take = [&](size_t bytes) -> char* {
        char* p = ws + off;
        off += (bytes + 255) & ~(size_t)255;
        return p;
    };
    float*    maxPart = (float*)take(256 * sizeof(float));
    float*    maxV    = (float*)take(256);
    _Float16* hxN     = (_Float16*)take((size_t)ROWS * KPAD * sizeof(_Float16));
    _Float16* aN0     = (_Float16*)take((size_t)ROWS * H * sizeof(_Float16));
    _Float16* aN1     = (_Float16*)take((size_t)ROWS * H * sizeof(_Float16));
    _Float16* aN2     = (_Float16*)take((size_t)ROWS * H * sizeof(_Float16));
    _Float16* wp1h    = (_Float16*)take((size_t)H * KPAD * sizeof(_Float16));
    _Float16* wq1h    = (_Float16*)take((size_t)H * H * sizeof(_Float16));
    _Float16* ws1h    = (_Float16*)take((size_t)H * H * sizeof(_Float16));
    _Float16* wp2h    = (_Float16*)take((size_t)H * H * sizeof(_Float16));
    _Float16* wq2h    = (_Float16*)take((size_t)H * H * sizeof(_Float16));
    _Float16* ws2h    = (_Float16*)take((size_t)H * H * sizeof(_Float16));
    _Float16* wp3h    = (_Float16*)take((size_t)H * H * sizeof(_Float16));
    _Float16* ws3h    = (_Float16*)take((size_t)H * H * sizeof(_Float16));
    float*    S1      = (float*)take((size_t)ROWS * H * sizeof(float));
    float*    S2      = (float*)take((size_t)ROWS * H * sizeof(float));
    float*    S3      = (float*)take((size_t)ROWS * H * sizeof(float));
    float*    T1      = (float*)take((size_t)ROWS * H * sizeof(float));
    float*    T2      = (float*)take((size_t)ROWS * H * sizeof(float));
    float*    T3      = (float*)take((size_t)ROWS * H * sizeof(float));

    const int TPB = 256;
    const int sqBlocks  = (H * H + TPB - 1) / TPB;
    const int preBlocks = (H * KPAD + TPB - 1) / TPB;

    // f16 weights (w1_pre padded 784 -> 800)
    wconv_kernel<<<preBlocks, TPB, 0, stream>>>(w1_pre,  wp1h, H, D_IN, KPAD);
    wconv_kernel<<<sqBlocks,  TPB, 0, stream>>>(w1_post, wq1h, H, H, H);
    wconv_kernel<<<sqBlocks,  TPB, 0, stream>>>(w1_self, ws1h, H, H, H);
    wconv_kernel<<<sqBlocks,  TPB, 0, stream>>>(w2_pre,  wp2h, H, H, H);
    wconv_kernel<<<sqBlocks,  TPB, 0, stream>>>(w2_post, wq2h, H, H, H);
    wconv_kernel<<<sqBlocks,  TPB, 0, stream>>>(w2_self, ws2h, H, H, H);
    wconv_kernel<<<sqBlocks,  TPB, 0, stream>>>(w3_pre,  wp3h, H, H, H);
    wconv_kernel<<<sqBlocks,  TPB, 0, stream>>>(w3_self, ws3h, H, H, H);

    // global max + overlay/normalize
    max1_kernel<<<256, TPB, 0, stream>>>(x, maxPart, BATCH * D_IN);
    max2_kernel<<<1,   TPB, 0, stream>>>(maxPart, maxV, 256);
    hx_kernel<<<ROWS,  TPB, 0, stream>>>(x, maxV, hxN);

    dim3 gG(H / 64, ROWS / 256);   // (32, 20): 8 waves/block, each wave = 32x64 tile

    // t = 0: states are zero -> post/self reduce to relu(bias) (K==0 path)
    meta_gemm_kernel<<<gG, TPB, 0, stream>>>(hxN, wp1h, b1_pre, KPAD,
                                             nullptr, nullptr, b1_post, 0,
                                             nullptr, nullptr, b1_self, 0, T1);
    norm_kernel<<<ROWS, TPB, 0, stream>>>(T1, aN0);
    meta_gemm_kernel<<<gG, TPB, 0, stream>>>(aN0, wp2h, b2_pre, H,
                                             nullptr, nullptr, b2_post, 0,
                                             nullptr, nullptr, b2_self, 0, T2);
    norm_kernel<<<ROWS, TPB, 0, stream>>>(T2, aN1);
    meta_gemm_kernel<<<gG, TPB, 0, stream>>>(aN1, wp3h, b3_pre, H,
                                             nullptr, nullptr, b3_post, 0,
                                             nullptr, nullptr, b3_self, 0, T3);

    float *s1p = T1, *s2p = T2, *s3p = T3, *t1p = S1, *t2p = S2, *t3p = S3;
    for (int t = 1; t < 3; ++t) {
        norm_kernel<<<ROWS, TPB, 0, stream>>>(s1p, aN0);
        norm_kernel<<<ROWS, TPB, 0, stream>>>(s2p, aN1);
        norm_kernel<<<ROWS, TPB, 0, stream>>>(s3p, aN2);
        meta_gemm_kernel<<<gG, TPB, 0, stream>>>(hxN, wp1h, b1_pre, KPAD,
                                                 aN1, wq1h, b1_post, H,
                                                 aN0, ws1h, b1_self, H, t1p);
        norm_kernel<<<ROWS, TPB, 0, stream>>>(t1p, aN0);   // n1 normalized (s1N dead)
        meta_gemm_kernel<<<gG, TPB, 0, stream>>>(aN0, wp2h, b2_pre, H,
                                                 aN2, wq2h, b2_post, H,
                                                 aN1, ws2h, b2_self, H, t2p);
        norm_kernel<<<ROWS, TPB, 0, stream>>>(t2p, aN1);   // n2 normalized (s2N dead)
        meta_gemm_kernel<<<gG, TPB, 0, stream>>>(aN1, wp3h, b3_pre, H,
                                                 nullptr, nullptr, b3_post, 0,
                                                 aN2, ws3h, b3_self, H, t3p);
        float* tmp;
        tmp = s1p; s1p = t1p; t1p = tmp;
        tmp = s2p; s2p = t2p; t2p = tmp;
        tmp = s3p; s3p = t3p; t3p = tmp;
    }

    goodness_kernel<<<ROWS, TPB, 0, stream>>>(s1p, s2p, s3p, out);
}
